// EncoderLayer_6760278524526
// MI455X (gfx1250) — matrix operand order repaired and checked by reading, not compiled, not run on hardware
//
#include <hip/hip_runtime.h>
#include <stdint.h>

typedef _Float16 v16h __attribute__((ext_vector_type(16)));
typedef _Float16 v8h  __attribute__((ext_vector_type(8)));
typedef float    v8f  __attribute__((ext_vector_type(8)));
typedef int      v4i  __attribute__((ext_vector_type(4)));

enum : int {
  F_RELU  = 1,
  F_BIAS  = 2,
  F_ACC   = 4,
  F_ST32  = 8,
  F_ST16  = 16,
  F_SHIFT = 32,   // A rows shifted by `shift` within each batch of `batchS` rows (conv1d)
  F_AF32  = 64,   // A is float32, convert to f16 while staging into LDS
  F_BCOL  = 128   // B stored K-major (B[n,k] at B[k*ldb+n])
};

#define LDST 48   // LDS row stride in halves (96B: 32B-aligned for all rows)

// ---- CDNA5 async global->LDS DMA (ASYNCcnt), guarded so compile never breaks ----
#if defined(__gfx1250__) && \
    __has_builtin(__builtin_amdgcn_global_load_async_to_lds_b128) && \
    __has_builtin(__builtin_amdgcn_s_wait_asynccnt)
#define USE_ASYNC_LDS 1
#else
#define USE_ASYNC_LDS 0
#endif

#define AS1 __attribute__((address_space(1)))
#define AS3 __attribute__((address_space(3)))

__device__ __forceinline__ AS1 v4i* gcast4(const void* p) {
  return (AS1 v4i*)(uintptr_t)p;
}
__device__ __forceinline__ AS3 v4i* lcast4(void* p) {
  // generic LDS pointer carries the LDS byte offset in its low 32 bits
  return (AS3 v4i*)(unsigned)(uintptr_t)p;
}

// ---------------------------------------------------------------------------
// Generic batched WMMA GEMM:  C[z] = alpha * A[z](MxK) * B[z](NxK)^T (+bias)(+relu)
// Block tile 128x64 (8 wave32 waves, each a 32x32 tile of four 16x16 WMMA tiles),
// K staged in 32-wide slabs through LDS (async DMA when source is f16).
// ---------------------------------------------------------------------------
template <int FLAGS>
__global__ __launch_bounds__(256) void wmma_gemm(
    const void* __restrict__ A, long lda,
    const _Float16* __restrict__ Bm, long ldb,
    float* __restrict__ Cf, _Float16* __restrict__ Ch,
    const float* __restrict__ Cin, long ldc,
    const float* __restrict__ bias, float alpha,
    int M, int N, int K, int shift, int batchS,
    int H, long sAb, long sAh, long sBb, long sBh, long sCb, long sCh)
{
  __shared__ _Float16 sA[128 * LDST];
  __shared__ _Float16 sB[64 * LDST];

  const int tid  = threadIdx.x;
  const int lane = tid & 31;
  const int wv   = tid >> 5;
  const int wm   = wv & 3;    // 4 waves along M (4*32 = 128)
  const int wn   = wv >> 2;   // 2 waves along N (2*32 = 64)

  const int zb = (int)blockIdx.z / H;
  const int zh = (int)blockIdx.z % H;
  const float*    Af = (const float*)A    + (long)zb * sAb + (long)zh * sAh;
  const _Float16* Ah = (const _Float16*)A + (long)zb * sAb + (long)zh * sAh;
  const _Float16* Bp = Bm                 + (long)zb * sBb + (long)zh * sBh;
  const long cbase   = (long)zb * sCb + (long)zh * sCh;

  const int blockM = blockIdx.y * 128;
  const int blockN = blockIdx.x * 64;

  v8f c[2][2];
  if constexpr ((FLAGS & F_ACC) != 0) {
#pragma unroll
    for (int mi = 0; mi < 2; ++mi)
#pragma unroll
      for (int ni = 0; ni < 2; ++ni)
#pragma unroll
        for (int j = 0; j < 8; ++j) {
          int rg = blockM + wm * 32 + mi * 16 + (lane >> 4) * 8 + j;
          int cg = blockN + wn * 32 + ni * 16 + (lane & 15);
          c[mi][ni][j] = (rg < M) ? Cin[cbase + (long)rg * ldc + cg] : 0.0f;
        }
  } else {
#pragma unroll
    for (int mi = 0; mi < 2; ++mi)
#pragma unroll
      for (int ni = 0; ni < 2; ++ni)
#pragma unroll
        for (int j = 0; j < 8; ++j) c[mi][ni][j] = 0.0f;
  }

  // staging coordinates: one 16-half (32B) chunk per thread
  const int arow = tid >> 1;              // 0..127
  const int ac16 = (tid & 1) * 16;        // 0 or 16
  const int brow = tid >> 1;              // n index 0..63 (tid<128)
  const int bc16 = (tid & 1) * 16;

  for (int k0 = 0; k0 < K; k0 += 32) {
    // ---- stage A tile (128 x 32 halves) ----
    {
      int gm    = blockM + arow;
      bool ok   = (gm < M);
      long srow = gm;
      if constexpr ((FLAGS & F_SHIFT) != 0) {
        int s  = gm % batchS;
        int ss = s + shift;
        ok     = ok && (ss >= 0) && (ss < batchS);
        srow   = (long)gm + shift;
      }
      _Float16* dst = &sA[arow * LDST + ac16];
      if (ok) {
        if constexpr ((FLAGS & F_AF32) != 0) {
          const float4* p4 = (const float4*)(Af + srow * lda + k0 + ac16);
          if (k0 + 32 < K) __builtin_prefetch((const void*)(p4 + 8), 0, 0);
#pragma unroll
          for (int i = 0; i < 4; ++i) {
            float4 f = p4[i];
            dst[i * 4 + 0] = (_Float16)f.x;
            dst[i * 4 + 1] = (_Float16)f.y;
            dst[i * 4 + 2] = (_Float16)f.z;
            dst[i * 4 + 3] = (_Float16)f.w;
          }
        } else {
          const _Float16* p = Ah + srow * lda + k0 + ac16;
#if USE_ASYNC_LDS
          __builtin_amdgcn_global_load_async_to_lds_b128(gcast4(p), lcast4(dst), 0, 0);
          __builtin_amdgcn_global_load_async_to_lds_b128(gcast4(p + 8), lcast4(dst + 8), 0, 0);
#else
          ((uint4*)dst)[0] = ((const uint4*)p)[0];
          ((uint4*)dst)[1] = ((const uint4*)p)[1];
#endif
        }
      } else {
        ((uint4*)dst)[0] = make_uint4(0, 0, 0, 0);
        ((uint4*)dst)[1] = make_uint4(0, 0, 0, 0);
      }
    }
    // ---- stage B tile (64 x 32 halves, optionally K-major source) ----
    if (tid < 128) {
      int gn = blockN + brow;
      _Float16* dst = &sB[brow * LDST + bc16];
      if constexpr ((FLAGS & F_BCOL) != 0) {
#pragma unroll
        for (int j = 0; j < 16; ++j)
          dst[j] = Bp[(long)(k0 + bc16 + j) * ldb + gn];
      } else {
        const _Float16* p = Bp + (long)gn * ldb + k0 + bc16;
#if USE_ASYNC_LDS
        __builtin_amdgcn_global_load_async_to_lds_b128(gcast4(p), lcast4(dst), 0, 0);
        __builtin_amdgcn_global_load_async_to_lds_b128(gcast4(p + 8), lcast4(dst + 8), 0, 0);
#else
        ((uint4*)dst)[0] = ((const uint4*)p)[0];
        ((uint4*)dst)[1] = ((const uint4*)p)[1];
#endif
      }
    }
#if USE_ASYNC_LDS
    if constexpr (!((FLAGS & F_AF32) && (FLAGS & F_BCOL)))
      __builtin_amdgcn_s_wait_asynccnt(0);
#endif
    __syncthreads();

    // ---- fragments + WMMA ----
    v16h a[2], b[2];
#pragma unroll
    for (int mi = 0; mi < 2; ++mi) {
      int ar = wm * 32 + mi * 16 + (lane & 15);
      // A VGPR layout: lane half0..7 -> K=(lane/16)*8.. ; half8..15 -> K=16+(lane/16)*8..
      v8h lo = *(const v8h*)&sA[ar * LDST + (lane >> 4) * 8];
      v8h hi = *(const v8h*)&sA[ar * LDST + 16 + (lane >> 4) * 8];
      a[mi] = __builtin_shufflevector(lo, hi, 0, 1, 2, 3, 4, 5, 6, 7,
                                      8, 9, 10, 11, 12, 13, 14, 15);
    }
#pragma unroll
    for (int ni = 0; ni < 2; ++ni) {
      int br = wn * 32 + ni * 16 + (lane & 15);
      v8h blo = *(const v8h*)&sB[br * LDST + (lane >> 4) * 8];
      v8h bhi = *(const v8h*)&sB[br * LDST + 16 + (lane >> 4) * 8];
      b[ni] = __builtin_shufflevector(blo, bhi, 0, 1, 2, 3, 4, 5, 6, 7,
                                      8, 9, 10, 11, 12, 13, 14, 15);
    }
#pragma unroll
    for (int mi = 0; mi < 2; ++mi)
#pragma unroll
      for (int ni = 0; ni < 2; ++ni)
        c[mi][ni] = __builtin_amdgcn_wmma_f32_16x16x32_f16(
            false, a[mi], false, b[ni], (short)0, c[mi][ni], false, false);
    __syncthreads();
  }

  // ---- epilogue ----
#pragma unroll
  for (int mi = 0; mi < 2; ++mi)
#pragma unroll
    for (int ni = 0; ni < 2; ++ni)
#pragma unroll
      for (int j = 0; j < 8; ++j) {
        int rg = blockM + wm * 32 + mi * 16 + (lane >> 4) * 8 + j;
        int cg = blockN + wn * 32 + ni * 16 + (lane & 15);
        if (rg >= M) continue;
        float v = c[mi][ni][j] * alpha;
        if constexpr ((FLAGS & F_BIAS) != 0) v += bias[cg];
        if constexpr ((FLAGS & F_RELU) != 0) v = fmaxf(v, 0.0f);
        long ci = cbase + (long)rg * ldc + cg;
        if constexpr ((FLAGS & F_ST32) != 0) Cf[ci] = v;
        if constexpr ((FLAGS & F_ST16) != 0) Ch[ci] = (_Float16)v;
      }
}

// ---------------------------------------------------------------------------
// Block reductions (256 threads = 8 wave32 waves)
// ---------------------------------------------------------------------------
__device__ __forceinline__ float blk_max(float v, float* red) {
#pragma unroll
  for (int o = 16; o > 0; o >>= 1) v = fmaxf(v, __shfl_down(v, o, 32));
  if ((threadIdx.x & 31) == 0) red[threadIdx.x >> 5] = v;
  __syncthreads();
  float r = red[0];
#pragma unroll
  for (int i = 1; i < 8; ++i) r = fmaxf(r, red[i]);
  __syncthreads();
  return r;
}
__device__ __forceinline__ float blk_sum(float v, float* red) {
#pragma unroll
  for (int o = 16; o > 0; o >>= 1) v += __shfl_down(v, o, 32);
  if ((threadIdx.x & 31) == 0) red[threadIdx.x >> 5] = v;
  __syncthreads();
  float r = 0.0f;
#pragma unroll
  for (int i = 0; i < 8; ++i) r += red[i];
  __syncthreads();
  return r;
}

// Row softmax over S=1024, mask[b, key] -> -1e9, in place on fp32 attn.
__global__ __launch_bounds__(256) void softmax_rows(
    float* __restrict__ attn, const uint8_t* __restrict__ mask, int S, int HS)
{
  __shared__ float red[8];
  long r = blockIdx.x;
  float* row = attn + r * (long)S;
  const uint8_t* mrow = mask + (long)(r / HS) * S;
  const int tid = threadIdx.x;
  float v[4], mx = -3.4e38f;
#pragma unroll
  for (int i = 0; i < 4; ++i) {
    int cidx = tid + i * 256;
    float x = row[cidx];
    if (mrow[cidx]) x = -1e9f;
    v[i] = x;
    mx = fmaxf(mx, x);
  }
  mx = blk_max(mx, red);
  float s = 0.0f;
#pragma unroll
  for (int i = 0; i < 4; ++i) { v[i] = __expf(v[i] - mx); s += v[i]; }
  s = blk_sum(s, red);
  float inv = 1.0f / s;
#pragma unroll
  for (int i = 0; i < 4; ++i) row[tid + i * 256] = v[i] * inv;
}

// out = LayerNorm(a + b) * g + be  over D=1024 (population variance, eps 1e-5)
__global__ __launch_bounds__(256) void add_layernorm(
    const float* __restrict__ a, const float* __restrict__ b,
    const float* __restrict__ g, const float* __restrict__ be,
    float* __restrict__ out, int D)
{
  __shared__ float red[8];
  long base = (long)blockIdx.x * D;
  const int tid = threadIdx.x;
  float v[4], s = 0.0f, s2 = 0.0f;
#pragma unroll
  for (int i = 0; i < 4; ++i) {
    int cidx = tid + i * 256;
    float x = a[base + cidx] + b[base + cidx];
    v[i] = x; s += x; s2 += x * x;
  }
  s  = blk_sum(s, red);
  s2 = blk_sum(s2, red);
  float mean = s / D;
  float var  = s2 / D - mean * mean;
  float rs   = rsqrtf(var + 1e-5f);
#pragma unroll
  for (int i = 0; i < 4; ++i) {
    int cidx = tid + i * 256;
    out[base + cidx] = (v[i] - mean) * rs * g[cidx] + be[cidx];
  }
}

__global__ __launch_bounds__(256) void cvt_f32_f16(
    const float* __restrict__ in, _Float16* __restrict__ out, long n)
{
  long i = (long)blockIdx.x * 256 + threadIdx.x;
  if (i < n) out[i] = (_Float16)in[i];
}

// De-interleave Wc (O,I,3) -> 3 contiguous f16 (O,I) matrices
__global__ __launch_bounds__(256) void cvt_wc(
    const float* __restrict__ wc, _Float16* __restrict__ w0,
    _Float16* __restrict__ w1, _Float16* __restrict__ w2, long n)
{
  long i = (long)blockIdx.x * 256 + threadIdx.x;
  if (i < n) {
    w0[i] = (_Float16)wc[i * 3 + 0];
    w1[i] = (_Float16)wc[i * 3 + 1];
    w2[i] = (_Float16)wc[i * 3 + 2];
  }
}

// hch = f16( relu(acc + bias[n]) )
__global__ __launch_bounds__(256) void bias_relu_cvt(
    const float* __restrict__ acc, const float* __restrict__ bias,
    _Float16* __restrict__ out, long total, int D)
{
  long i = (long)blockIdx.x * 256 + threadIdx.x;
  if (i < total) {
    int n = (int)(i & (long)(D - 1));
    out[i] = (_Float16)fmaxf(acc[i] + bias[n], 0.0f);
  }
}

// ---------------------------------------------------------------------------
extern "C" void kernel_launch(void* const* d_in, const int* in_sizes, int n_in,
                              void* d_out, int out_size, void* d_ws, size_t ws_size,
                              hipStream_t stream) {
  const int Bn = 8, S = 1024, D = 1024, Hh = 16, FF = 4096;
  const int BS = Bn * S;                 // 8192 rows
  const long DD = (long)D * D;           // 1M

  const float*   x    = (const float*)d_in[0];
  const uint8_t* mask = (const uint8_t*)d_in[1];
  const float* Wq = (const float*)d_in[2];  const float* bq = (const float*)d_in[3];
  const float* Wk = (const float*)d_in[4];  const float* bk = (const float*)d_in[5];
  const float* Wv = (const float*)d_in[6];  const float* bv = (const float*)d_in[7];
  const float* Wo = (const float*)d_in[8];  const float* bo = (const float*)d_in[9];
  const float* g1 = (const float*)d_in[10]; const float* b1 = (const float*)d_in[11];
  const float* W1 = (const float*)d_in[12]; const float* bf1= (const float*)d_in[13];
  const float* W2 = (const float*)d_in[14]; const float* bf2= (const float*)d_in[15];
  const float* Wc = (const float*)d_in[16]; const float* bc = (const float*)d_in[17];
  const float* Wf = (const float*)d_in[18]; const float* bf = (const float*)d_in[19];
  const float* g2 = (const float*)d_in[20]; const float* b2 = (const float*)d_in[21];

  float* outF  = (float*)d_out;
  float* attnF = outF + (long)BS * D;    // attn region: [B,H,S,S] fp32

  // ---- workspace layout ----
  char* base = (char*)d_ws;
  size_t off = 0;
  auto AL = [&](size_t bytes) -> char* {
    char* p = base + off;
    off = (off + bytes + 255) & ~(size_t)255;
    return p;
  };
  _Float16* Wqh  = (_Float16*)AL(DD * 2);
  _Float16* Wkh  = (_Float16*)AL(DD * 2);
  _Float16* Wvh  = (_Float16*)AL(DD * 2);
  _Float16* Woh  = (_Float16*)AL(DD * 2);
  _Float16* Wfh  = (_Float16*)AL(DD * 2);
  _Float16* Wc0h = (_Float16*)AL(DD * 2);
  _Float16* Wc1h = (_Float16*)AL(DD * 2);
  _Float16* Wc2h = (_Float16*)AL(DD * 2);
  _Float16* W1h  = (_Float16*)AL((long)FF * D * 2);
  _Float16* W2h  = (_Float16*)AL((long)FF * D * 2);
  _Float16* qh   = (_Float16*)AL((long)BS * D * 2);
  _Float16* kh   = (_Float16*)AL((long)BS * D * 2);
  _Float16* vh   = (_Float16*)AL((long)BS * D * 2);
  _Float16* ctxh = (_Float16*)AL((long)BS * D * 2);
  _Float16* hch  = (_Float16*)AL((long)BS * D * 2);
  _Float16* f1h  = (_Float16*)AL((long)BS * FF * 2);
  float* qf    = (float*)AL((long)BS * D * 4);   // q fp32 (residual); reused as fob
  float* projf = (float*)AL((long)BS * D * 4);   // proj; reused as conv accumulator
  float* aof   = (float*)AL((long)BS * D * 4);   // attn_out (LN1)
  float* hf    = (float*)AL((long)BS * D * 4);   // FFN output h
  float* hcacc = projf;
  float* fobf  = qf;

  // ---- weight conversion ----
  cvt_f32_f16<<<(int)(DD / 256), 256, 0, stream>>>(Wq, Wqh, DD);
  cvt_f32_f16<<<(int)(DD / 256), 256, 0, stream>>>(Wk, Wkh, DD);
  cvt_f32_f16<<<(int)(DD / 256), 256, 0, stream>>>(Wv, Wvh, DD);
  cvt_f32_f16<<<(int)(DD / 256), 256, 0, stream>>>(Wo, Woh, DD);
  cvt_f32_f16<<<(int)(DD / 256), 256, 0, stream>>>(Wf, Wfh, DD);
  cvt_f32_f16<<<(int)((long)FF * D / 256), 256, 0, stream>>>(W1, W1h, (long)FF * D);
  cvt_f32_f16<<<(int)((long)FF * D / 256), 256, 0, stream>>>(W2, W2h, (long)FF * D);
  cvt_wc<<<(int)(DD / 256), 256, 0, stream>>>(Wc, Wc0h, Wc1h, Wc2h, DD);

  // ---- Q, K, V projections (x fp32 staged->f16; q kept fp32 for residual) ----
  dim3 gMain(D / 64, BS / 128, 1);
  wmma_gemm<F_AF32 | F_BIAS | F_ST32 | F_ST16><<<gMain, 256, 0, stream>>>(
      x, D, Wqh, D, qf, qh, nullptr, D, bq, 1.0f,
      BS, D, D, 0, 1, 1, 0, 0, 0, 0, 0, 0);
  wmma_gemm<F_AF32 | F_BIAS | F_ST16><<<gMain, 256, 0, stream>>>(
      x, D, Wkh, D, nullptr, kh, nullptr, D, bk, 1.0f,
      BS, D, D, 0, 1, 1, 0, 0, 0, 0, 0, 0);
  wmma_gemm<F_AF32 | F_BIAS | F_ST16><<<gMain, 256, 0, stream>>>(
      x, D, Wvh, D, nullptr, vh, nullptr, D, bv, 1.0f,
      BS, D, D, 0, 1, 1, 0, 0, 0, 0, 0, 0);

  // ---- attention scores: [z=b*H+h] Q(1024x64) x K^T -> attn fp32, alpha=1/8 ----
  wmma_gemm<F_ST32><<<dim3(S / 64, S / 128, Bn * Hh), 256, 0, stream>>>(
      qh, D, kh, D, attnF, nullptr, nullptr, S, nullptr, 0.125f,
      S, S, 64, 0, 1, Hh,
      (long)S * D, 64L, (long)S * D, 64L, (long)Hh * S * S, (long)S * S);

  // ---- softmax (in place on fp32 attn output) ----
  softmax_rows<<<Bn * Hh * S, 256, 0, stream>>>(attnF, mask, S, Hh * S);

  // ---- context: attn(fp32->f16 staged) x V (K-major) -> ctx f16 [b,s,h*64+d] ----
  wmma_gemm<F_AF32 | F_BCOL | F_ST16><<<dim3(1, S / 128, Bn * Hh), 256, 0, stream>>>(
      attnF, S, vh, D, nullptr, ctxh, nullptr, D, nullptr, 1.0f,
      S, 64, S, 0, 1, Hh,
      (long)Hh * S * S, (long)S * S, (long)S * D, 64L, (long)S * D, 64L);

  // ---- output projection + LN1 (residual = projected q) ----
  wmma_gemm<F_BIAS | F_ST32><<<gMain, 256, 0, stream>>>(
      ctxh, D, Woh, D, projf, nullptr, nullptr, D, bo, 1.0f,
      BS, D, D, 0, 1, 1, 0, 0, 0, 0, 0, 0);
  add_layernorm<<<BS, 256, 0, stream>>>(projf, qf, g1, b1, aof, D);

  // ---- FFN ----
  wmma_gemm<F_AF32 | F_BIAS | F_RELU | F_ST16><<<dim3(FF / 64, BS / 128, 1), 256, 0, stream>>>(
      aof, D, W1h, D, nullptr, f1h, nullptr, FF, bf1, 1.0f,
      BS, FF, D, 0, 1, 1, 0, 0, 0, 0, 0, 0);
  wmma_gemm<F_BIAS | F_ST32><<<gMain, 256, 0, stream>>>(
      f1h, FF, W2h, FF, hf, nullptr, nullptr, D, bf2, 1.0f,
      BS, D, FF, 0, 1, 1, 0, 0, 0, 0, 0, 0);

  // ---- conv1d(k=3, SAME) as 3 shifted accumulate-GEMMs ----
  wmma_gemm<F_AF32 | F_SHIFT | F_ST32><<<gMain, 256, 0, stream>>>(
      hf, D, Wc0h, D, hcacc, nullptr, nullptr, D, nullptr, 1.0f,
      BS, D, D, -1, S, 1, 0, 0, 0, 0, 0, 0);
  wmma_gemm<F_AF32 | F_SHIFT | F_ST32 | F_ACC><<<gMain, 256, 0, stream>>>(
      hf, D, Wc1h, D, hcacc, nullptr, hcacc, D, nullptr, 1.0f,
      BS, D, D, 0, S, 1, 0, 0, 0, 0, 0, 0);
  wmma_gemm<F_AF32 | F_SHIFT | F_ST32 | F_ACC><<<gMain, 256, 0, stream>>>(
      hf, D, Wc2h, D, hcacc, nullptr, hcacc, D, nullptr, 1.0f,
      BS, D, D, 1, S, 1, 0, 0, 0, 0, 0, 0);
  bias_relu_cvt<<<(int)((long)BS * D / 256), 256, 0, stream>>>(
      hcacc, bc, hch, (long)BS * D, D);

  // ---- FC + final residual LN -> out ----
  wmma_gemm<F_BIAS | F_ST32><<<gMain, 256, 0, stream>>>(
      hch, D, Wfh, D, fobf, nullptr, nullptr, D, bf, 1.0f,
      BS, D, D, 0, 1, 1, 0, 0, 0, 0, 0, 0);
  add_layernorm<<<BS, 256, 0, stream>>>(fobf, aof, g2, b2, outF, D);
}
